// TSGCNext_unit_82145544503457
// MI455X (gfx1250) — compile-verified, matrix operand order checked by reading, not run on hardware
//
#include <hip/hip_runtime.h>
#include <hip/hip_bf16.h>
#include <math.h>

// ---------------------------------------------------------------------------
// Types for CDNA5 WMMA (wave32): v_wmma_f32_16x16x32_bf16
// ---------------------------------------------------------------------------
typedef __bf16 v16bf __attribute__((ext_vector_type(16)));
typedef __bf16 v8bf  __attribute__((ext_vector_type(8)));
typedef float  v8f   __attribute__((ext_vector_type(8)));

__device__ __forceinline__ v8f wmma_bf16(v16bf a, v16bf b, v8f c) {
    // 8 args: (neg_a, A, neg_b, B, c_mod, C, reuse_a, reuse_b)
    return __builtin_amdgcn_wmma_f32_16x16x32_bf16(false, a, false, b,
                                                   (short)0, c, false, false);
}

// Build a 16-bit A fragment (16x32 tile) from LDS per ISA layout:
// lane L: row M = mt*16 + (L&15); group g = L>>4;
// element e -> K = kb*32 + ((e&7) | ((e>>3)<<4) | (g<<3))
// => two contiguous 8-element (16B) LDS loads at k0 and k0+16.
__device__ __forceinline__ v16bf load_afrag(const __bf16* base, int stride,
                                            int lane, int mt, int kb) {
    int row = mt * 16 + (lane & 15);
    int k0  = kb * 32 + ((lane >> 4) << 3);
    const __bf16* p = base + row * stride + k0;
    v8bf lo = *(const v8bf*)(p);
    v8bf hi = *(const v8bf*)(p + 16);
    return __builtin_shufflevector(lo, hi,
        0, 1, 2, 3, 4, 5, 6, 7, 8, 9, 10, 11, 12, 13, 14, 15);
}

__device__ __forceinline__ float gelu_exact(float s) {
    return 0.5f * s * (1.0f + erff(s * 0.70710678118654752f));
}

// ---------------------------------------------------------------------------
// Prep: swizzle w1 (128x512) and w2 (512x128) into bf16 B-fragment layout,
// and copy PA (1875 f32) into the output tail (reference returns (y, PA)).
// B-fragment layout (mirrors A with N in place of M):
//   flat = ((kb*NT + nt)*32 + lane)*16 + e
//   col  = nt*16 + (lane&15)
//   k    = kb*32 + ((e&7) | ((e>>3)<<4) | ((lane>>4)<<3))
// ---------------------------------------------------------------------------
__global__ __launch_bounds__(256) void prep_kernel(
    const float* __restrict__ w1, const float* __restrict__ w2,
    const float* __restrict__ PA,
    __bf16* __restrict__ w1s, __bf16* __restrict__ w2s,
    float* __restrict__ pa_out)
{
    int tid = blockIdx.x * 256 + threadIdx.x;
    if (tid < 65536) {                       // w1s: kb 0..3, nt 0..31
        int o = tid;
        int e = o & 15, L = (o >> 4) & 31, tile = o >> 9;
        int nt = tile & 31, kb = tile >> 5;
        int k = kb * 32 + ((e & 7) | ((e >> 3) << 4) | ((L >> 4) << 3));
        int n = nt * 16 + (L & 15);
        w1s[o] = (__bf16)w1[k * 512 + n];
    } else if (tid < 131072) {               // w2s: kb 0..15, nt 0..7
        int o = tid - 65536;
        int e = o & 15, L = (o >> 4) & 31, tile = o >> 9;
        int nt = tile & 7, kb = tile >> 3;
        int k = kb * 32 + ((e & 7) | ((e >> 3) << 4) | ((L >> 4) << 3));
        int n = nt * 16 + (L & 15);
        w2s[o] = (__bf16)w2[k * 128 + n];
    } else if (tid < 131072 + 1875) {        // PA copy to output tail
        int o = tid - 131072;
        pa_out[o] = PA[o];
    }
}

// ---------------------------------------------------------------------------
// Fused main kernel: one block per (n,t); 256 threads = 8 wave32.
// conv(T,k=3) -> LN(C) -> GEMM1 (WMMA bf16) -> PA mix -> GELU ->
// GEMM2 (WMMA bf16) -> +x residual.
// ---------------------------------------------------------------------------
__global__ __launch_bounds__(256) void tsg_kernel(
    const float* __restrict__ x,
    const float* __restrict__ dw_w, const float* __restrict__ dw_b,
    const float* __restrict__ ln_w, const float* __restrict__ ln_b,
    const __bf16* __restrict__ w1s, const float* __restrict__ b1,
    const float* __restrict__ PA,
    const __bf16* __restrict__ w2s, const float* __restrict__ b2,
    float* __restrict__ y)
{
    constexpr int C = 128, T = 128, V = 25;
    constexpr int SH_STR = 132;              // padded f32 row stride

    __shared__ __bf16 sA[32 * 128];          //  8 KB : LN'd activations (GEMM1 A)
    __shared__ __bf16 sB1[32 * 512];         // 32 KB : GEMM1 out -> mixed/GELU (GEMM2 A)
    __shared__ float  sH[V * SH_STR];        // 13 KB : conv out, later GEMM2 out
    __shared__ float  sPA[3 * 25 * 25];      // 7.5KB : PA staged

    const int tid  = threadIdx.x;
    const int lane = tid & 31;
    const int wave = tid >> 5;
    const int n    = blockIdx.x / T;
    const int t    = blockIdx.x % T;

    // ---- step 0: stage PA, zero A-tile pad rows (25..31) --------------------
    for (int i = tid; i < 1875; i += 256) sPA[i] = PA[i];
    for (int i = tid; i < 7 * 128; i += 256) sA[25 * 128 + i] = (__bf16)0.0f;

    // ---- step 1: depthwise conv over T (k=3, pad 1) -------------------------
    for (int i = tid; i < C * V; i += 256) {
        int c = i / V, v = i % V;
        float acc = dw_b[c];
        float k0 = dw_w[c * 3 + 0], k1 = dw_w[c * 3 + 1], k2 = dw_w[c * 3 + 2];
        long base = ((long)(n * C + c) * T) * V + v;
        if (t > 0)     acc += k0 * x[base + (long)(t - 1) * V];
        acc += k1 * x[base + (long)t * V];
        if (t < T - 1) acc += k2 * x[base + (long)(t + 1) * V];
        sH[v * SH_STR + c] = acc;
    }
    __syncthreads();

    // ---- step 2: LayerNorm over C, one wave per v-row, write bf16 A --------
    for (int v = wave; v < V; v += 8) {
        float s1 = 0.f, s2 = 0.f, vals[4];
        for (int j = 0; j < 4; ++j) {
            float h = sH[v * SH_STR + lane + j * 32];
            vals[j] = h; s1 += h; s2 += h * h;
        }
        for (int m = 16; m >= 1; m >>= 1) {
            s1 += __shfl_xor(s1, m);
            s2 += __shfl_xor(s2, m);
        }
        float mu   = s1 * (1.f / 128.f);
        float var  = s2 * (1.f / 128.f) - mu * mu;
        float rstd = rsqrtf(var + 1e-6f);
        for (int j = 0; j < 4; ++j) {
            int c = lane + j * 32;
            sA[v * 128 + c] = (__bf16)((vals[j] - mu) * rstd * ln_w[c] + ln_b[c]);
        }
    }
    __syncthreads();

    // ---- step 3: GEMM1 [32x128]x[128x512], wave owns 2 Mt x 4 Nt -----------
    {
        v8f acc[2][4] = {};
        for (int kb = 0; kb < 4; ++kb) {
            v16bf a0 = load_afrag(sA, 128, lane, 0, kb);
            v16bf a1 = load_afrag(sA, 128, lane, 1, kb);
            for (int j = 0; j < 4; ++j) {
                int nt = wave * 4 + j;
                v16bf b = *(const v16bf*)(w1s + (size_t)((kb * 32 + nt) * 32 + lane) * 16);
                acc[0][j] = wmma_bf16(a0, b, acc[0][j]);
                acc[1][j] = wmma_bf16(a1, b, acc[1][j]);
            }
        }
        // C layout: lane L, reg r -> row = mt*16 + 8*(L>=16) + r, col = Nt*16 + (L&15)
        for (int j = 0; j < 4; ++j) {
            int col  = (wave * 4 + j) * 16 + (lane & 15);
            float bias = b1[col];
            for (int m = 0; m < 2; ++m) {
                int rowbase = m * 16 + ((lane >> 4) << 3);
                for (int r = 0; r < 8; ++r)
                    sB1[(rowbase + r) * 512 + col] = (__bf16)(acc[m][j][r] + bias);
            }
        }
    }
    __syncthreads();

    // ---- step 4: PA graph-mix over V (cols d%4<3) + exact GELU, in place ---
    for (int d = tid; d < 512; d += 256) {
        int i = d & 3;
        float col[25];
        for (int k = 0; k < 25; ++k) col[k] = (float)sB1[k * 512 + d];
        if (i < 3) {
            const float* pa = &sPA[i * 625];
            for (int u = 0; u < 25; ++u) {
                float s = 0.f;
                for (int k = 0; k < 25; ++k) s += pa[u * 25 + k] * col[k];
                sB1[u * 512 + d] = (__bf16)gelu_exact(s);
            }
        } else {
            for (int u = 0; u < 25; ++u)
                sB1[u * 512 + d] = (__bf16)gelu_exact(col[u]);
        }
    }
    __syncthreads();

    // ---- step 5: GEMM2 [32x512]x[512x128], wave owns 2 Mt x 1 Nt -----------
    {
        v8f acc[2] = {};
        for (int kb = 0; kb < 16; ++kb) {
            v16bf a0 = load_afrag(sB1, 512, lane, 0, kb);
            v16bf a1 = load_afrag(sB1, 512, lane, 1, kb);
            v16bf b  = *(const v16bf*)(w2s + (size_t)((kb * 8 + wave) * 32 + lane) * 16);
            acc[0] = wmma_bf16(a0, b, acc[0]);
            acc[1] = wmma_bf16(a1, b, acc[1]);
        }
        int c = wave * 16 + (lane & 15);
        for (int m = 0; m < 2; ++m) {
            int rowbase = m * 16 + ((lane >> 4) << 3);
            for (int r = 0; r < 8; ++r) {
                int v = rowbase + r;
                if (v < V) sH[v * SH_STR + c] = acc[m][r];   // pad rows discarded
            }
        }
    }
    __syncthreads();

    // ---- step 6: residual epilogue y = x + h + b2 --------------------------
    for (int i = tid; i < C * V; i += 256) {
        int c = i / V, v = i % V;
        long gi = ((long)(n * C + c) * T + t) * V + v;
        y[gi] = x[gi] + sH[v * SH_STR + c] + b2[c];
    }
}

// ---------------------------------------------------------------------------
// Launch
// ---------------------------------------------------------------------------
extern "C" void kernel_launch(void* const* d_in, const int* in_sizes, int n_in,
                              void* d_out, int out_size, void* d_ws, size_t ws_size,
                              hipStream_t stream) {
    const float* x    = (const float*)d_in[0];
    const float* dw_w = (const float*)d_in[1];
    const float* dw_b = (const float*)d_in[2];
    const float* ln_w = (const float*)d_in[3];
    const float* ln_b = (const float*)d_in[4];
    const float* w1   = (const float*)d_in[5];
    const float* b1   = (const float*)d_in[6];
    const float* PA   = (const float*)d_in[7];
    const float* w2   = (const float*)d_in[8];
    const float* b2   = (const float*)d_in[9];

    float* out = (float*)d_out;
    __bf16* w1s = (__bf16*)d_ws;           // 65536 bf16 = 128 KB
    __bf16* w2s = w1s + 65536;             // 65536 bf16 = 128 KB
    float* pa_out = out + (out_size - 1875);

    // 65536 + 65536 + 1875 = 132947 work items
    prep_kernel<<<(132947 + 255) / 256, 256, 0, stream>>>(w1, w2, PA, w1s, w2s, pa_out);

    // one block per (n,t): 64*128 = 8192 blocks of 256 threads (8 wave32)
    tsg_kernel<<<8192, 256, 0, stream>>>(x, dw_w, dw_b, ln_w, ln_b,
                                         w1s, b1, PA, w2s, b2, out);
}
